// MSTParserLSTMModel_14568529068133
// MI455X (gfx1250) — compile-verified, matrix operand order checked by reading, not run on hardware
//
#include <hip/hip_runtime.h>
#include <hip/hip_bf16.h>

typedef __attribute__((ext_vector_type(16))) _Float16 v16h;
typedef __attribute__((ext_vector_type(8)))  _Float16 v8h;
typedef __attribute__((ext_vector_type(8)))  float    v8f;

#define SEQ_N 768
#define LSTM_H 125
#define GATES 500    /* 4*LSTM_H */
#define FEAT_D 175
#define FEAT_KP 192  /* 175 padded to mult of 32 */
#define CAT_D 250
#define CAT_KP 256   /* 250 padded to mult of 32 */
#define HID_U 100

__device__ __forceinline__ float sigmoid_f(float x) {
    return 1.0f / (1.0f + __expf(-x));
}
__device__ __forceinline__ float tanh_f(float x) {
    return 2.0f / (1.0f + __expf(-2.0f * x)) - 1.0f;  // hardware v_exp_f32
}

// ---------------------------------------------------------------------------
// 1) Embedding gather -> f16 packed activations feats[768][192] (K zero-pad)
// ---------------------------------------------------------------------------
__global__ void gather_feats(const int* __restrict__ wid, const int* __restrict__ pid,
                             const int* __restrict__ oid, const int* __restrict__ cid,
                             const float* __restrict__ wemb, const float* __restrict__ pemb,
                             const float* __restrict__ oemb, const float* __restrict__ cemb,
                             _Float16* __restrict__ featsH) {
    int t = blockIdx.x;
    int j = threadIdx.x;            // 0..191
    float v = 0.0f;
    if (j < 100)      v = wemb[wid[t] * 100 + j];
    else if (j < 125) v = pemb[pid[t] * 25 + (j - 100)];
    else if (j < 150) v = oemb[oid[t] * 25 + (j - 125)];
    else if (j < 175) v = cemb[cid[t] * 25 + (j - 150)];
    featsH[(size_t)t * FEAT_KP + j] = (_Float16)v;
}

// ---------------------------------------------------------------------------
// 2) Weight pack: src -> f16 [Np][Kp], zero padded.
//    transSrc=0: src is [N,K] (weight layout). transSrc=1: src is [K,N].
// ---------------------------------------------------------------------------
__global__ void pack_weight(const float* __restrict__ src, _Float16* __restrict__ dst,
                            int N, int K, int Np, int Kp, int transSrc) {
    int idx = blockIdx.x * blockDim.x + threadIdx.x;
    if (idx >= Np * Kp) return;
    int n = idx / Kp, k = idx % Kp;
    float v = 0.0f;
    if (n < N && k < K)
        v = transSrc ? src[(size_t)k * N + n] : src[(size_t)n * K + k];
    dst[idx] = (_Float16)v;
}

__global__ void fill_zero_h(_Float16* __restrict__ p, int n) {
    int i = blockIdx.x * blockDim.x + threadIdx.x;
    if (i < n) p[i] = (_Float16)0.0f;
}

// ---------------------------------------------------------------------------
// 3) Transpose Whh [500,125] -> WhhT [125,500] for coalesced recurrence reads
// ---------------------------------------------------------------------------
__global__ void transpose_mat(const float* __restrict__ src, float* __restrict__ dst,
                              int R, int C) {
    int idx = blockIdx.x * blockDim.x + threadIdx.x;
    if (idx < R * C) {
        int r = idx / C, c = idx % C;
        dst[c * R + r] = src[idx];
    }
}

// ---------------------------------------------------------------------------
// 4) Register-direct WMMA GEMM on packed f16:
//      D[M,N] (f32) = A16[M,Kp] * B16[Np,Kp]^T  (+bias0[n]) (+bias1[n])
//    One wave per 16(M) x 32(N) tile: A fragment reused across 2 accumulators.
//    Fragment K-mapping: lane half hi -> K runs {8hi..8hi+7} and {8hi+16..8hi+23}
//    = two contiguous 16-byte chunks -> two b128 loads per fragment.
// ---------------------------------------------------------------------------
__global__ void gemm_wmma_packed(const _Float16* __restrict__ A16,
                                 const _Float16* __restrict__ B16,
                                 const float* __restrict__ bias0,
                                 const float* __restrict__ bias1,
                                 float* __restrict__ D,
                                 int M, int N, int Kp) {
    const int lane = threadIdx.x;       // 0..31
    const int m0 = blockIdx.y * 16;
    const int n0 = blockIdx.x * 32;
    const int hi = lane >> 4;
    const int mn = lane & 15;

    v8f acc0 = {0.f, 0.f, 0.f, 0.f, 0.f, 0.f, 0.f, 0.f};
    v8f acc1 = {0.f, 0.f, 0.f, 0.f, 0.f, 0.f, 0.f, 0.f};

    const _Float16* __restrict__ arow  = A16 + (size_t)(m0 + mn) * Kp + 8 * hi;
    const _Float16* __restrict__ brow0 = B16 + (size_t)(n0 + mn) * Kp + 8 * hi;
    const _Float16* __restrict__ brow1 = B16 + (size_t)(n0 + 16 + mn) * Kp + 8 * hi;

    for (int k0 = 0; k0 < Kp; k0 += 32) {
        v8h alo = *(const v8h*)(arow + k0);
        v8h ahi = *(const v8h*)(arow + k0 + 16);
        v8h b0l = *(const v8h*)(brow0 + k0);
        v8h b0h = *(const v8h*)(brow0 + k0 + 16);
        v8h b1l = *(const v8h*)(brow1 + k0);
        v8h b1h = *(const v8h*)(brow1 + k0 + 16);
        v16h af  = __builtin_shufflevector(alo, ahi, 0,1,2,3,4,5,6,7,8,9,10,11,12,13,14,15);
        v16h bf0 = __builtin_shufflevector(b0l, b0h, 0,1,2,3,4,5,6,7,8,9,10,11,12,13,14,15);
        v16h bf1 = __builtin_shufflevector(b1l, b1h, 0,1,2,3,4,5,6,7,8,9,10,11,12,13,14,15);
        acc0 = __builtin_amdgcn_wmma_f32_16x16x32_f16(false, af, false, bf0,
                                                      (short)0, acc0, false, false);
        acc1 = __builtin_amdgcn_wmma_f32_16x16x32_f16(false, af, false, bf1,
                                                      (short)0, acc1, false, false);
    }

    // D layout: VGPR r -> row M = r + 8*hi, column N = lane%16 (+0 / +16)
#pragma unroll
    for (int r = 0; r < 8; ++r) {
        int gm = m0 + r + hi * 8;
        int gn = n0 + mn;
        float b = 0.0f;
        if (gn < N) {
            if (bias0) b += bias0[gn];
            if (bias1) b += bias1[gn];
            D[(size_t)gm * N + gn] = acc0[r] + b;
        }
        int gn1 = gn + 16;
        if (gn1 < N) {
            float b1 = 0.0f;
            if (bias0) b1 += bias0[gn1];
            if (bias1) b1 += bias1[gn1];
            D[(size_t)gm * N + gn1] = acc1[r] + b1;
        }
    }
}

// ---------------------------------------------------------------------------
// 5) Sequential LSTM scan. grid.x=2 (fwd/bwd), block=512.
//    z_j = Zx[t,j] + sum_k WhhT[k,j]*h[k]; gates i,f,g,o.
//    Writes hidden state directly into packed f16 activation buffer
//    out16[t*outStride + off + j] (pad columns pre-zeroed).
// ---------------------------------------------------------------------------
__global__ void lstm_scan(const float* __restrict__ ZxF, const float* __restrict__ ZxB,
                          const float* __restrict__ WhhTF, const float* __restrict__ WhhTB,
                          _Float16* __restrict__ out16, int outStride, int offF, int offB,
                          int n) {
    const int dir = blockIdx.x;
    const float* __restrict__ Zx   = dir ? ZxB   : ZxF;
    const float* __restrict__ WhhT = dir ? WhhTB : WhhTF;
    const int off = dir ? offB : offF;

    __shared__ float h_s[LSTM_H];
    __shared__ float c_s[LSTM_H];
    __shared__ float z_s[GATES];

    const int j = threadIdx.x;
    if (j < LSTM_H) { h_s[j] = 0.0f; c_s[j] = 0.0f; }
    __syncthreads();

    for (int t = 0; t < n; ++t) {
        const int tt = dir ? (n - 1 - t) : t;
        if (j < GATES) {
            float z = Zx[(size_t)tt * GATES + j];
#pragma unroll 5
            for (int k = 0; k < LSTM_H; ++k)
                z = fmaf(WhhT[k * GATES + j], h_s[k], z);
            z_s[j] = z;
        }
        __syncthreads();
        if (j < LSTM_H) {
            float ig = sigmoid_f(z_s[j]);
            float fg = sigmoid_f(z_s[LSTM_H + j]);
            float gg = tanh_f(z_s[2 * LSTM_H + j]);
            float og = sigmoid_f(z_s[3 * LSTM_H + j]);
            float c  = fg * c_s[j] + ig * gg;
            float hn = og * tanh_f(c);
            c_s[j] = c;
            h_s[j] = hn;
            out16[(size_t)tt * outStride + off + j] = (_Float16)hn;
        }
        __syncthreads();
    }
}

// ---------------------------------------------------------------------------
// 6) Pairwise scoring: scores[i,j] = sum_h tanh(head[i,h]+mod[j,h])*outW[h]+outB
//    (hidBias folded into headfov). 59M hardware-exp tanh evaluations.
// ---------------------------------------------------------------------------
__global__ void pairwise_scores(const float* __restrict__ headf,
                                const float* __restrict__ modf,
                                const float* __restrict__ outW,
                                const float* __restrict__ outB,
                                float* __restrict__ scores, int n) {
    __shared__ float hh[16][HID_U];
    __shared__ float mm[16][HID_U];
    __shared__ float ww[HID_U];

    const int i0 = blockIdx.y * 16;
    const int j0 = blockIdx.x * 16;
    const int tid = threadIdx.y * 16 + threadIdx.x;

    for (int idx = tid; idx < 16 * HID_U; idx += 256) {
        int r = idx / HID_U, c = idx % HID_U;
        hh[r][c] = headf[(size_t)(i0 + r) * HID_U + c];
        mm[r][c] = modf[(size_t)(j0 + r) * HID_U + c];
    }
    if (tid < HID_U) ww[tid] = outW[tid];
    __syncthreads();

    const int ty = threadIdx.y, tx = threadIdx.x;
    float s = outB[0];
#pragma unroll 4
    for (int h = 0; h < HID_U; ++h)
        s = fmaf(tanh_f(hh[ty][h] + mm[tx][h]), ww[h], s);
    scores[(size_t)(i0 + ty) * n + (j0 + tx)] = s;
}

// ---------------------------------------------------------------------------
// Launch
// ---------------------------------------------------------------------------
extern "C" void kernel_launch(void* const* d_in, const int* in_sizes, int n_in,
                              void* d_out, int out_size, void* d_ws, size_t ws_size,
                              hipStream_t stream) {
    const int*   wid     = (const int*)d_in[0];
    const int*   pid     = (const int*)d_in[1];
    const int*   oid     = (const int*)d_in[2];
    const int*   cid     = (const int*)d_in[3];
    const float* wemb    = (const float*)d_in[4];
    const float* pemb    = (const float*)d_in[5];
    const float* oemb    = (const float*)d_in[6];
    const float* cemb    = (const float*)d_in[7];
    const float* Wih_f1  = (const float*)d_in[8];
    const float* Whh_f1  = (const float*)d_in[9];
    const float* bih_f1  = (const float*)d_in[10];
    const float* bhh_f1  = (const float*)d_in[11];
    const float* Wih_b1  = (const float*)d_in[12];
    const float* Whh_b1  = (const float*)d_in[13];
    const float* bih_b1  = (const float*)d_in[14];
    const float* bhh_b1  = (const float*)d_in[15];
    const float* Wih_f2  = (const float*)d_in[16];
    const float* Whh_f2  = (const float*)d_in[17];
    const float* bih_f2  = (const float*)d_in[18];
    const float* bhh_f2  = (const float*)d_in[19];
    const float* Wih_b2  = (const float*)d_in[20];
    const float* Whh_b2  = (const float*)d_in[21];
    const float* bih_b2  = (const float*)d_in[22];
    const float* bhh_b2  = (const float*)d_in[23];
    const float* WFOH    = (const float*)d_in[24];
    const float* WFOM    = (const float*)d_in[25];
    const float* hidBias = (const float*)d_in[26];
    const float* outW    = (const float*)d_in[27];
    const float* outB    = (const float*)d_in[28];
    float*       scores  = (float*)d_out;
    (void)ws_size; (void)n_in; (void)in_sizes; (void)out_size;

    // --- Workspace layout: f32 region first, then 16B-aligned f16 region ---
    float* F = (float*)d_ws;
    size_t o = 0;
    float* zxf1    = F + o; o += (size_t)SEQ_N * GATES;   // 768x500
    float* zxb1    = F + o; o += (size_t)SEQ_N * GATES;
    float* zxf2    = F + o; o += (size_t)SEQ_N * GATES;
    float* zxb2    = F + o; o += (size_t)SEQ_N * GATES;
    float* headf   = F + o; o += (size_t)SEQ_N * HID_U;   // 768x100
    float* modf    = F + o; o += (size_t)SEQ_N * HID_U;
    float* whhT_f1 = F + o; o += (size_t)GATES * LSTM_H;  // 125x500 each
    float* whhT_b1 = F + o; o += (size_t)GATES * LSTM_H;
    float* whhT_f2 = F + o; o += (size_t)GATES * LSTM_H;
    float* whhT_b2 = F + o; o += (size_t)GATES * LSTM_H;

    _Float16* Hbase = (_Float16*)(F + o);
    size_t h = 0;
    _Float16* featsH = Hbase + h; h += (size_t)SEQ_N * FEAT_KP;  // 768x192
    _Float16* w1fH   = Hbase + h; h += (size_t)512 * FEAT_KP;    // 512x192
    _Float16* w1bH   = Hbase + h; h += (size_t)512 * FEAT_KP;
    _Float16* w2fH   = Hbase + h; h += (size_t)512 * CAT_KP;     // 512x256
    _Float16* w2bH   = Hbase + h; h += (size_t)512 * CAT_KP;
    _Float16* wfohH  = Hbase + h; h += (size_t)128 * CAT_KP;     // 128x256
    _Float16* wfomH  = Hbase + h; h += (size_t)128 * CAT_KP;
    _Float16* cat1H  = Hbase + h; h += (size_t)SEQ_N * CAT_KP;   // 768x256
    _Float16* HbufH  = Hbase + h; h += (size_t)SEQ_N * CAT_KP;

    // 1) Embedding gather -> packed f16 feats
    gather_feats<<<SEQ_N, FEAT_KP, 0, stream>>>(wid, pid, oid, cid,
                                                wemb, pemb, oemb, cemb, featsH);

    // 2) Zero padded activation buffers (pad columns must be 0 for WMMA K)
    {
        int nh = SEQ_N * CAT_KP;
        fill_zero_h<<<(nh + 255) / 256, 256, 0, stream>>>(cat1H, nh);
        fill_zero_h<<<(nh + 255) / 256, 256, 0, stream>>>(HbufH, nh);
    }

    // 3) Pack weights to f16 [Np][Kp]
    {
        int n1 = 512 * FEAT_KP, n2 = 512 * CAT_KP, n3 = 128 * CAT_KP;
        pack_weight<<<(n1 + 255) / 256, 256, 0, stream>>>(Wih_f1, w1fH, GATES, FEAT_D, 512, FEAT_KP, 0);
        pack_weight<<<(n1 + 255) / 256, 256, 0, stream>>>(Wih_b1, w1bH, GATES, FEAT_D, 512, FEAT_KP, 0);
        pack_weight<<<(n2 + 255) / 256, 256, 0, stream>>>(Wih_f2, w2fH, GATES, CAT_D, 512, CAT_KP, 0);
        pack_weight<<<(n2 + 255) / 256, 256, 0, stream>>>(Wih_b2, w2bH, GATES, CAT_D, 512, CAT_KP, 0);
        pack_weight<<<(n3 + 255) / 256, 256, 0, stream>>>(WFOH, wfohH, HID_U, CAT_D, 128, CAT_KP, 1);
        pack_weight<<<(n3 + 255) / 256, 256, 0, stream>>>(WFOM, wfomH, HID_U, CAT_D, 128, CAT_KP, 1);
    }

    // 4) Transpose Whh matrices for the recurrence
    {
        const int tg = (GATES * LSTM_H + 255) / 256;
        transpose_mat<<<tg, 256, 0, stream>>>(Whh_f1, whhT_f1, GATES, LSTM_H);
        transpose_mat<<<tg, 256, 0, stream>>>(Whh_b1, whhT_b1, GATES, LSTM_H);
        transpose_mat<<<tg, 256, 0, stream>>>(Whh_f2, whhT_f2, GATES, LSTM_H);
        transpose_mat<<<tg, 256, 0, stream>>>(Whh_b2, whhT_b2, GATES, LSTM_H);
    }

    // 5) Layer-1 input projections: Zx = feats @ Wih^T + bih + bhh
    {
        dim3 grid((GATES + 31) / 32, SEQ_N / 16);   // 16 x 48
        gemm_wmma_packed<<<grid, 32, 0, stream>>>(featsH, w1fH, bih_f1, bhh_f1,
                                                  zxf1, SEQ_N, GATES, FEAT_KP);
        gemm_wmma_packed<<<grid, 32, 0, stream>>>(featsH, w1bH, bih_b1, bhh_b1,
                                                  zxb1, SEQ_N, GATES, FEAT_KP);
    }

    // 6) Layer-1 bi-LSTM -> cat1H (f16, [768][256], cols 0..124 fwd, 125..249 bwd)
    lstm_scan<<<2, 512, 0, stream>>>(zxf1, zxb1, whhT_f1, whhT_b1,
                                     cat1H, CAT_KP, 0, LSTM_H, SEQ_N);

    // 7) Layer-2 input projections (K = 256 padded)
    {
        dim3 grid((GATES + 31) / 32, SEQ_N / 16);
        gemm_wmma_packed<<<grid, 32, 0, stream>>>(cat1H, w2fH, bih_f2, bhh_f2,
                                                  zxf2, SEQ_N, GATES, CAT_KP);
        gemm_wmma_packed<<<grid, 32, 0, stream>>>(cat1H, w2bH, bih_b2, bhh_b2,
                                                  zxb2, SEQ_N, GATES, CAT_KP);
    }

    // 8) Layer-2 bi-LSTM -> HbufH (f16 packed)
    lstm_scan<<<2, 512, 0, stream>>>(zxf2, zxb2, whhT_f2, whhT_b2,
                                     HbufH, CAT_KP, 0, LSTM_H, SEQ_N);

    // 9) headfov = H @ WFOH + hidBias ; modfov = H @ WFOM
    {
        dim3 grid((HID_U + 31) / 32, SEQ_N / 16);   // 4 x 48
        gemm_wmma_packed<<<grid, 32, 0, stream>>>(HbufH, wfohH, hidBias, nullptr,
                                                  headf, SEQ_N, HID_U, CAT_KP);
        gemm_wmma_packed<<<grid, 32, 0, stream>>>(HbufH, wfomH, nullptr, nullptr,
                                                  modf, SEQ_N, HID_U, CAT_KP);
    }

    // 10) Pairwise tanh scoring -> scores [768,768]
    {
        dim3 grid(SEQ_N / 16, SEQ_N / 16);
        dim3 block(16, 16);
        pairwise_scores<<<grid, block, 0, stream>>>(headf, modf, outW, outB,
                                                    scores, SEQ_N);
    }
}
